// SocialPoolingModule_12283606468138
// MI455X (gfx1250) — compile-verified, matrix operand order checked by reading, not run on hardware
//
#include <hip/hip_runtime.h>

typedef __attribute__((ext_vector_type(16))) _Float16 v16h;
typedef __attribute__((ext_vector_type(8)))  _Float16 v8h;
typedef __attribute__((ext_vector_type(8)))  float    v8f;

// ---------------- constants ----------------
#define BB 16
#define NN 128
#define HH 128
#define BD 64
#define PP 256

// Packed WMMA operand tile: 16x32 f16 tile stored as [lane 0..31][16 halves]
// (1024 bytes). Lane L's 16 operand halves are contiguous -> 2x ds_load_b128.
// Mapping (ISA 7.12.2, 16-bit A/B): slot s (0..15), half h = L>>4:
//   ko = 8*h + (s & 7) + 16*(s >> 3)        (K offset within 32-K block)
// inverse: h = (ko>>3)&1 ; s = (ko&7) + 8*(ko>>4)
// A-tile: lane coord r = L&15 = row within 16-row M tile.
// B-tile: lane coord n = L&15 = col within 16-col N tile.
// C/D f32 fragment: col n = L&15, row m = v + 8*(L>>4).
// C-packed 16x16 f32 tile in memory: tile*256 + lane*8 + v  (32B/lane).

static __device__ __forceinline__ v16h ld_tile(const _Float16* tileBase) {
  int lane = threadIdx.x & 31;
  const v8h* p = (const v8h*)(tileBase + lane * 16);
  union { v16h v; v8h h[2]; } u;
  u.h[0] = p[0];
  u.h[1] = p[1];
  return u.v;
}

static __device__ __forceinline__ v8f wmma_(v16h a, v16h b, v8f c) {
  return __builtin_amdgcn_wmma_f32_16x16x32_f16(false, a, false, b, (short)0, c, false, false);
}

static __device__ __forceinline__ int ko_of(int h, int s) {
  return 8 * h + (s & 7) + 16 * (s >> 3);
}

// Dual-accumulator, software-pipelined (depth-1) K loop over packed B tiles.
// Next K-step's loads are issued before the current WMMAs.
template <int KT>
static __device__ __forceinline__ void gemm_pair(const _Float16* Bp, const v16h (&aT)[KT],
                                                 int nt0, int nt1, v8f& acc0, v8f& acc1) {
  v16h b0 = ld_tile(Bp + (0 * 16 + nt0) * 512);
  v16h b1 = ld_tile(Bp + (0 * 16 + nt1) * 512);
#pragma unroll
  for (int kk = 0; kk < KT; ++kk) {
    v16h c0 = b0, c1 = b1;
    if (kk + 1 < KT) {
      b0 = ld_tile(Bp + ((kk + 1) * 16 + nt0) * 512);
      b1 = ld_tile(Bp + ((kk + 1) * 16 + nt1) * 512);
    }
    acc0 = wmma_(aT[kk], c0, acc0);
    acc1 = wmma_(aT[kk], c1, acc1);
  }
}

// ---------------- kernel 1: stage weights/hidden into packed WMMA layouts ---------
__global__ void prep_kernel(const float* __restrict__ hs, const float* __restrict__ W2,
                            const float* __restrict__ W3, const float* __restrict__ Wout,
                            _Float16* hsA, _Float16* W2hB, _Float16* W2eB,
                            _Float16* W3B, _Float16* WoutB) {
  int g = blockIdx.x * 256 + threadIdx.x;
  if (g < BB * NN * HH) {            // hsA (A-ready, per b): tiles (mt 0..7, kb 0..3)
    int b = g >> 14, o = g & 16383;
    int s = o & 15, L = (o >> 4) & 31, tile = o >> 9;
    int r = L & 15, h = L >> 4;
    int mt = tile >> 2, kb = tile & 3;
    hsA[g] = (_Float16)hs[(size_t)(b * NN + mt * 16 + r) * HH + kb * 32 + ko_of(h, s)];
  }
  int g2 = g - BB * NN * HH;
  if (g2 >= 0 && g2 < HH * PP) {     // W2hB (rows 0..127 of W2): tiles (kb, nt)
    int s = g2 & 15, L = (g2 >> 4) & 31, tile = g2 >> 9;
    int n = L & 15, h = L >> 4;
    int kb = tile >> 4, nt = tile & 15;
    W2hB[g2] = (_Float16)W2[(size_t)(kb * 32 + ko_of(h, s)) * PP + nt * 16 + n];
  }
  int g3 = g2 - HH * PP;
  if (g3 >= 0 && g3 < BD * PP) {     // W2eB (rows 128..191 of W2)
    int s = g3 & 15, L = (g3 >> 4) & 31, tile = g3 >> 9;
    int n = L & 15, h = L >> 4;
    int kb = tile >> 4, nt = tile & 15;
    W2eB[g3] = (_Float16)W2[(size_t)(HH + kb * 32 + ko_of(h, s)) * PP + nt * 16 + n];
  }
  int g4 = g3 - BD * PP;
  if (g4 >= 0 && g4 < PP * PP) {     // W3B and WoutB
    int s = g4 & 15, L = (g4 >> 4) & 31, tile = g4 >> 9;
    int n = L & 15, h = L >> 4;
    int kb = tile >> 4, nt = tile & 15;
    size_t src = (size_t)(kb * 32 + ko_of(h, s)) * PP + nt * 16 + n;
    W3B[g4]   = (_Float16)W3[src];
    WoutB[g4] = (_Float16)Wout[src];
  }
}

// ---------------- kernel 2: hsW2C = C-packed (hs @ W2[0:128,:] + b2) ----------
// hsW2C tile index = (b*8 + mt)*16 + nt ; element = tile*256 + lane*8 + v.
__global__ void hsw2_kernel(const _Float16* __restrict__ hsA, const _Float16* __restrict__ W2hB,
                            const float* __restrict__ b2, float* __restrict__ hsW2C) {
  extern __shared__ char smem[];
  _Float16* Ap = (_Float16*)smem;          // 32 tiles  (16384 halves)
  _Float16* Bp = Ap + 32 * 512;            // 64 tiles  (32768 halves)
  int b = blockIdx.x, t = threadIdx.x;
  const v8h* ga = (const v8h*)(hsA + (size_t)b * NN * HH);
  const v8h* gb = (const v8h*)W2hB;
  for (int idx = t; idx < (NN * HH) / 8; idx += 256) ((v8h*)Ap)[idx] = ga[idx];
  for (int idx = t; idx < (HH * PP) / 8; idx += 256) ((v8h*)Bp)[idx] = gb[idx];
  __syncthreads();
  int wave = t >> 5, lane = t & 31, n = lane & 15;
  int mt = wave;                           // 8 waves x 16 n-tiles
  v16h aT[4];
#pragma unroll
  for (int kk = 0; kk < 4; ++kk) aT[kk] = ld_tile(Ap + (mt * 4 + kk) * 512);
#pragma unroll
  for (int p = 0; p < 8; ++p) {            // nt pairs
    int nt0 = 2 * p, nt1 = 2 * p + 1;
    v8f acc0 = {}, acc1 = {};
    gemm_pair<4>(Bp, aT, nt0, nt1, acc0, acc1);
#pragma unroll
    for (int tn = 0; tn < 2; ++tn) {
      int nt = 2 * p + tn;
      float bias = b2[nt * 16 + n];
      v8f res;
      const v8f& acc = tn ? acc1 : acc0;
#pragma unroll
      for (int v = 0; v < 8; ++v) res[v] = acc[v] + bias;
      size_t tile = (size_t)(b * 8 + mt) * 16 + nt;
      *(v8f*)(hsW2C + tile * 256 + lane * 8) = res;   // 32B/lane contiguous
    }
  }
}

// ---------------- kernel 3: fused per-(b,i) pipeline with on-the-fly max pool ----
__global__ void social_main_kernel(const float* __restrict__ pos, const float* __restrict__ W1,
                                   const float* __restrict__ b1, const float* __restrict__ b3,
                                   const float* __restrict__ hsW2C,
                                   const _Float16* __restrict__ W2eB,
                                   const _Float16* __restrict__ W3B,
                                   _Float16* __restrict__ pooledA) {
  extern __shared__ char smem[];
  _Float16* W3p = (_Float16*)smem;         // 128 tiles (65536 h) resident
  _Float16* W2p = W3p + 128 * 512;         // 32 tiles  (16384 h) resident
  _Float16* Y1p = W2p + 32 * 512;          // 16 tiles  (8192 h)  per-chunk A
  _Float16* Esp = Y1p + 16 * 512;          // 4 tiles   (2048 h)  per-chunk A

  int t = threadIdx.x, wave = t >> 5, lane = t & 31, n = lane & 15, hh = lane >> 4;
  int b  = blockIdx.x >> 4;
  int i0 = (blockIdx.x & 15) * 8;
  int nt0 = wave * 2, nt1 = wave * 2 + 1;
  int col0 = nt0 * 16 + n, col1 = nt1 * 16 + n;

  for (int idx = t; idx < (PP * PP) / 8; idx += 256) ((v8h*)W3p)[idx] = ((const v8h*)W3B)[idx];
  for (int idx = t; idx < (BD * PP) / 8; idx += 256) ((v8h*)W2p)[idx] = ((const v8h*)W2eB)[idx];
  __syncthreads();

  for (int ii = 0; ii < 8; ++ii) {
    int i = i0 + ii;
    float pix = pos[(b * NN + i) * 2 + 0];
    float piy = pos[(b * NN + i) * 2 + 1];
    float pmax0 = -3.0e38f, pmax1 = -3.0e38f;

    for (int jc = 0; jc < 4; ++jc) {       // j chunks of 32
      int jbase = jc * 32;
      { // enc chunk: relu((pos_j - pos_i) @ W1 + b1) -> packed A layout, 1 b128 store
        int jl = t >> 3, ks = (t & 7) * 8;
        int gj = b * NN + jbase + jl;
        float dx = pos[gj * 2 + 0] - pix;
        float dy = pos[gj * 2 + 1] - piy;
        v8h ev;
#pragma unroll
        for (int kk = 0; kk < 8; ++kk) {
          int k = ks + kk;
          float e = dx * W1[k] + dy * W1[BD + k] + b1[k];
          ev[kk] = (_Float16)(e > 0.f ? e : 0.f);
        }
        int mt = jl >> 4, r = jl & 15;
        int kb = ks >> 5, ko0 = ks & 31;
        int h2 = (ko0 >> 3) & 1, s0 = 8 * (ko0 >> 4);
        *(v8h*)(Esp + (mt * 2 + kb) * 512 + (r + 16 * h2) * 16 + s0) = ev;
      }
      __syncthreads();                     // Esp ready; protects Y1p reuse

      // GEMM1: Y1[32x256] = relu(enc @ W2e + hsW2), write packed-A layout
#pragma unroll
      for (int mt = 0; mt < 2; ++mt) {
        // C-packed hsW2 prefetch: one 32B load per (lane, nt)
        size_t mtile = (size_t)(b * 8 + jc * 2 + mt) * 16;
        v8f hv0 = *(const v8f*)(hsW2C + (mtile + nt0) * 256 + lane * 8);
        v8f hv1 = *(const v8f*)(hsW2C + (mtile + nt1) * 256 + lane * 8);
        v16h aT[2];
        aT[0] = ld_tile(Esp + (mt * 2 + 0) * 512);
        aT[1] = ld_tile(Esp + (mt * 2 + 1) * 512);
        v8f acc0 = {}, acc1 = {};
        gemm_pair<2>(W2p, aT, nt0, nt1, acc0, acc1);
#pragma unroll
        for (int tn = 0; tn < 2; ++tn) {
          const v8f& acc = tn ? acc1 : acc0;
          const v8f& hv  = tn ? hv1 : hv0;
          int col = tn ? col1 : col0;
          int kb = col >> 5, ko = col & 31;
          int h2 = (ko >> 3) & 1, s = (ko & 7) + 8 * (ko >> 4);
          _Float16* yb = Y1p + (mt * 8 + kb) * 512 + 16 * h2 * 16 + s;
#pragma unroll
          for (int v = 0; v < 8; ++v) {
            float y = acc[v] + hv[v];      // b2 folded into hsW2
            yb[(v + 8 * hh) * 16] = (_Float16)(y > 0.f ? y : 0.f);
          }
        }
      }
      __syncthreads();                     // Y1p ready

      // GEMM2: emb[32x256] = Y1 @ W3 ; masked max into registers
#pragma unroll
      for (int mt = 0; mt < 2; ++mt) {
        v16h aT[8];
#pragma unroll
        for (int kk = 0; kk < 8; ++kk) aT[kk] = ld_tile(Y1p + (mt * 8 + kk) * 512);
        v8f acc0 = {}, acc1 = {};
        gemm_pair<8>(W3p, aT, nt0, nt1, acc0, acc1);
        float m0 = -3.0e38f, m1 = -3.0e38f;
#pragma unroll
        for (int v = 0; v < 8; ++v) {
          int j = jbase + mt * 16 + v + 8 * hh;
          bool self = (j == i);
          m0 = fmaxf(m0, self ? -3.0e38f : acc0[v]);
          m1 = fmaxf(m1, self ? -3.0e38f : acc1[v]);
        }
        pmax0 = fmaxf(pmax0, m0);
        pmax1 = fmaxf(pmax1, m1);
      }
    }

    // combine half-wave row groups; store pooled in A-ready global layout
#pragma unroll
    for (int tn = 0; tn < 2; ++tn) {
      float pm = tn ? pmax1 : pmax0;
      pm = fmaxf(pm, __shfl_xor(pm, 16, 32));
      if (lane < 16) {
        int col = (wave * 2 + tn) * 16 + lane;
        int row = b * NN + i;
        int mtg = row >> 4, r = row & 15;
        int kb = col >> 5, ko = col & 31;
        int h2 = (ko >> 3) & 1, s = (ko & 7) + 8 * (ko >> 4);
        pooledA[(size_t)(mtg * 8 + kb) * 512 + (r + 16 * h2) * 16 + s] =
            (_Float16)(pm + b3[col]);
      }
    }
  }
}

// ---------------- kernel 4: out = pooled @ Wout + bout ----------------
__global__ void out_kernel(const _Float16* __restrict__ pooledA,
                           const _Float16* __restrict__ WoutB,
                           const float* __restrict__ bout, float* __restrict__ out) {
  extern __shared__ char smem[];
  _Float16* Ap = (_Float16*)smem;          // 32 tiles (16384 h) : 64 rows
  _Float16* Bp = Ap + 32 * 512;            // 128 tiles (65536 h)
  int t = threadIdx.x, wave = t >> 5, lane = t & 31, n = lane & 15, hh = lane >> 4;
  int row0 = blockIdx.x * 64;              // 32 blocks cover 2048 rows
  const v8h* ga = (const v8h*)(pooledA + (size_t)row0 * PP);  // tiles are mt-major
  for (int idx = t; idx < (64 * PP) / 8; idx += 256) ((v8h*)Ap)[idx] = ga[idx];
  for (int idx = t; idx < (PP * PP) / 8; idx += 256) ((v8h*)Bp)[idx] = ((const v8h*)WoutB)[idx];
  __syncthreads();
  int mt = wave >> 1;                      // 4 m-tiles x 16 n-tiles / 8 waves
  v16h aT[8];
#pragma unroll
  for (int kk = 0; kk < 8; ++kk) aT[kk] = ld_tile(Ap + (mt * 8 + kk) * 512);
#pragma unroll
  for (int p = 0; p < 4; ++p) {            // 8 nt per wave, processed in pairs
    int nt0 = (wave & 1) * 8 + 2 * p, nt1 = nt0 + 1;
    v8f acc0 = {}, acc1 = {};
    gemm_pair<8>(Bp, aT, nt0, nt1, acc0, acc1);
#pragma unroll
    for (int tn = 0; tn < 2; ++tn) {
      const v8f& acc = tn ? acc1 : acc0;
      int col = (tn ? nt1 : nt0) * 16 + n;
      float bias = bout[col];
#pragma unroll
      for (int v = 0; v < 8; ++v)
        out[((size_t)(row0 + mt * 16 + v + 8 * hh)) * PP + col] = acc[v] + bias;
    }
  }
}

// ---------------- launcher ----------------
extern "C" void kernel_launch(void* const* d_in, const int* in_sizes, int n_in,
                              void* d_out, int out_size, void* d_ws, size_t ws_size,
                              hipStream_t stream) {
  const float* hs   = (const float*)d_in[0];
  const float* pos  = (const float*)d_in[1];
  const float* W1   = (const float*)d_in[2];
  const float* b1   = (const float*)d_in[3];
  const float* W2   = (const float*)d_in[4];
  const float* b2   = (const float*)d_in[5];
  const float* W3   = (const float*)d_in[6];
  const float* b3   = (const float*)d_in[7];
  const float* Wout = (const float*)d_in[8];
  const float* bout = (const float*)d_in[9];

  char* ws = (char*)d_ws;
  float*    hsW2C   = (float*)(ws + 0);                 // 2 MB, C-packed (b2 folded)
  _Float16* pooledA = (_Float16*)(ws + 2097152);        // 1 MB (A-ready)
  _Float16* hsA     = (_Float16*)(ws + 3145728);        // 512 KB
  _Float16* W2hB    = (_Float16*)(ws + 3670016);        // 64 KB
  _Float16* W2eB    = (_Float16*)(ws + 3735552);        // 32 KB
  _Float16* W3B     = (_Float16*)(ws + 3768320);        // 128 KB
  _Float16* WoutB   = (_Float16*)(ws + 3899392);        // 128 KB

  const int SMEM_HSW2 = (32 + 64) * 512 * 2;            //  98304 B
  const int SMEM_MAIN = (128 + 32 + 16 + 4) * 512 * 2;  // 184320 B
  const int SMEM_OUT  = (32 + 128) * 512 * 2;           // 163840 B

  // CDNA5 WGP has 320 KB LDS; opt in to >64KB dynamic LDS.
  hipFuncSetAttribute((const void*)hsw2_kernel,        hipFuncAttributeMaxDynamicSharedMemorySize, SMEM_HSW2);
  hipFuncSetAttribute((const void*)social_main_kernel, hipFuncAttributeMaxDynamicSharedMemorySize, SMEM_MAIN);
  hipFuncSetAttribute((const void*)out_kernel,         hipFuncAttributeMaxDynamicSharedMemorySize, SMEM_OUT);

  prep_kernel<<<1728, 256, 0, stream>>>(hs, W2, W3, Wout, hsA, W2hB, W2eB, W3B, WoutB);
  hsw2_kernel<<<BB, 256, SMEM_HSW2, stream>>>(hsA, W2hB, b2, hsW2C);
  social_main_kernel<<<BB * (NN / 8), 256, SMEM_MAIN, stream>>>(pos, W1, b1, b3, hsW2C, W2eB, W3B, pooledA);
  out_kernel<<<32, 256, SMEM_OUT, stream>>>(pooledA, WoutB, bout, (float*)d_out);
}
